// EncoderLayer_10617159156311
// MI455X (gfx1250) — matrix operand order repaired and checked by reading, not compiled, not run on hardware
//
#include <hip/hip_runtime.h>

// ============================================================================
// MatNet-style bipartite encoder layer for MI455X (gfx1250, wave32, WMMA).
//
// v3: cost matrix pre-transposed once so both encoder blocks stream it with
// compile-time (512,1) strides -> clause-able immediate-offset loads instead
// of a serial 64-bit address chain.  All WMMA operands pre-packed f16
// (weights transposed, V transposed to [b][h][d][c]); flash-style fused
// attention; f32 accumulate everywhere.
// ============================================================================

typedef _Float16 half16 __attribute__((ext_vector_type(16)));
typedef _Float16 half8  __attribute__((ext_vector_type(8)));
typedef float    float8 __attribute__((ext_vector_type(8)));

#define SHUF16(a, b) __builtin_shufflevector(a, b, 0, 1, 2, 3, 4, 5, 6, 7, 8, 9, 10, 11, 12, 13, 14, 15)

// Geometry is fixed by the harness: B=2, R=C=512, E=256, H=16, D=16, F=512.
#define SEQ 512
#define EMB 256

// ---------------------------------------------------------------------------
// Packing kernels (one pass per launch; operands are re-read 16-64x by the
// GEMMs / attention, so converting to f16 + friendly layout once is a win).
// ---------------------------------------------------------------------------
__global__ void __launch_bounds__(256)
pack_f16(const float* __restrict__ X, _Float16* __restrict__ Y, int n)
{
  const int i = blockIdx.x * 256 + threadIdx.x;
  if (i < n) Y[i] = (_Float16)X[i];
}

// B[K,N] f32 -> Bt[N,K] f16 (weight transpose-pack)
__global__ void __launch_bounds__(256)
pack_bt_f16(const float* __restrict__ B, _Float16* __restrict__ Bt, int K, int N)
{
  const int i = blockIdx.x * 256 + threadIdx.x;
  if (i >= K * N) return;
  const int k = i / N, n = i - k * N;
  Bt[(size_t)n * K + k] = (_Float16)B[i];
}

// V (B,C,H*16+d) f32 -> Vt (B,H,16,C) f16 so attention B-fragments are contiguous
__global__ void __launch_bounds__(256)
pack_vt_f16(const float* __restrict__ V, _Float16* __restrict__ Vt, int Cn)
{
  const int i = blockIdx.x * 256 + threadIdx.x;   // i = (b*Cn + c)*256 + h*16 + d
  if (i >= 2 * Cn * 256) return;
  const int d = i & 15, h = (i >> 4) & 15;
  const int bc = i >> 8;
  const int c = bc % Cn, b = bc / Cn;
  Vt[((size_t)((b * 16 + h) * 16 + d)) * Cn + c] = (_Float16)V[i];
}

// 512x512 per-batch transpose (LDS 32x33 tile, coalesced both directions)
__global__ void __launch_bounds__(256)
transpose512(const float* __restrict__ X, float* __restrict__ Y)
{
  __shared__ float t[32][33];
  const int b  = blockIdx.z;
  const int i0 = blockIdx.y * 32, j0 = blockIdx.x * 32;
  const int tx = threadIdx.x & 31, ty = threadIdx.x >> 5;   // 32 x 8
  const float* xb = X + (size_t)b * SEQ * SEQ;
  float*       yb = Y + (size_t)b * SEQ * SEQ;
#pragma unroll
  for (int r = ty; r < 32; r += 8)
    t[r][tx] = xb[(size_t)(j0 + r) * SEQ + i0 + tx];
  __syncthreads();
#pragma unroll
  for (int r = ty; r < 32; r += 8)
    yb[(size_t)(i0 + r) * SEQ + j0 + tx] = t[tx][r];
}

// ---------------------------------------------------------------------------
// WMMA GEMM: C[M,N] = epilogue(A[M,K] @ Bt^T (+bias) (, Res))
// A: f16 row-major [M,K]; Bt: f16 [N,K] (pre-transposed weights).
// MODE 0 = none, 1 = relu, 2 = +Res, 3 = store f16 (C is _Float16*).
// One wave per 16x16 tile, 8 waves/block; every fragment load is b128.
// ---------------------------------------------------------------------------
template<int MODE>
__global__ void __launch_bounds__(256)
gemm16_wmma(const _Float16* __restrict__ A, const _Float16* __restrict__ Bt,
            const float* __restrict__ bias, const float* __restrict__ Res,
            float* __restrict__ C, int M, int N, int K)
{
  const int lane   = threadIdx.x & 31;
  const int wid    = threadIdx.x >> 5;
  const int ntiles = N >> 4;
  const int mtiles = M >> 4;
  const int tile   = blockIdx.x * 8 + wid;
  if (tile >= mtiles * ntiles) return;           // wave-uniform: EXEC stays full
  const int mt = tile / ntiles;
  const int nt = tile - mt * ntiles;

  const int mrow = lane & 15;                    // A-frag row
  const int ksel = lane >> 4;                    // K sub-block select (0/1)
  const int ncol = nt * 16 + (lane & 15);        // B/C-frag column
  const int kb   = ksel * 8;                     // B-frag K base

  const _Float16* arow = A  + (size_t)(mt * 16 + mrow) * K + ksel * 8;
  const _Float16* brow = Bt + (size_t)ncol * K + kb;

  float8 acc = {};
  for (int k0 = 0; k0 < K; k0 += 32) {
    if (k0 + 32 < K) {
      __builtin_prefetch(arow + k0 + 32, 0, 3);  // global_prefetch_b8
      __builtin_prefetch(brow + k0 + 32, 0, 3);
    }
    const half8 a0 = *(const half8*)(arow + k0);       // K = ksel*8 + 0..7
    const half8 a1 = *(const half8*)(arow + k0 + 16);  // K = 16 + ksel*8 + 0..7
    const half8 b0 = *(const half8*)(brow + k0);       // K = ksel*8 + 0..7
    const half8 b1 = *(const half8*)(brow + k0 + 16);  // K = 16 + ksel*8 + 0..7
    const half16 af = SHUF16(a0, a1);
    const half16 bf = SHUF16(b0, b1);
    acc = __builtin_amdgcn_wmma_f32_16x16x32_f16(false, af, false, bf,
                                                 (short)0, acc, false, false);
  }

#pragma unroll
  for (int i = 0; i < 8; ++i) {
    const int row = mt * 16 + i + ksel * 8;      // C/D layout: rows i + 8*(lane>>4)
    float v = acc[i];
    if (bias) v += bias[ncol];
    if (MODE == 1) v = v > 0.f ? v : 0.f;
    if (MODE == 2) v += Res[(size_t)row * N + ncol];
    if (MODE == 3) ((_Float16*)C)[(size_t)row * N + ncol] = (_Float16)v;
    else           C[(size_t)row * N + ncol] = v;
  }
}

// ---------------------------------------------------------------------------
// Fused mixed-score flash attention (geometry hardcoded: 512x512, H=16, D=16).
// One wave per (b, h, 16-row tile).  Streams 16-wide column tiles:
//   s  = (Q @ K^T) / sqrt(D)        (WMMA, K padded 16->32; Kh is f16)
//   s  = mixer MLP(s, cost)         (f32 VALU, 2->16->1 ReLU MLP)
//   online softmax (running max/sum per row, 16-lane shfl_xor reductions)
//   O += P @ V                      (WMMA; P via LDS D->A relayout; Vt f16)
// cost is pre-oriented row-major (stride 512,1) -> immediate-offset loads.
// ---------------------------------------------------------------------------
__global__ void __launch_bounds__(32)
attn_mixed_flash(const float* __restrict__ Q, const _Float16* __restrict__ Kh,
                 const _Float16* __restrict__ Vt, const float* __restrict__ cost,
                 const float* __restrict__ mix1_w, const float* __restrict__ mix1_b,
                 const float* __restrict__ mix2_w, const float* __restrict__ mix2_b,
                 float* __restrict__ Out)
{
  __shared__ _Float16 pl[16][16];                // P-tile relayout buffer (512 B)
  const int lane = threadIdx.x & 31;
  const int rt = blockIdx.x & 31;                // SEQ/16 = 32 row tiles
  const int bh = blockIdx.x >> 5;
  const int h  = bh & 15;                        // H = 16
  const int b  = bh >> 4;
  const int r0 = rt * 16;
  const int lm = lane & 15;
  const int g  = lane >> 4;                      // half-wave group (0/1)

  // per-head mixer weights (lane-uniform -> scalarized by compiler)
  float w1a[16], w1b[16], b1v[16], w2v[16];
#pragma unroll
  for (int j = 0; j < 16; ++j) {
    w1a[j] = mix1_w[h * 32 + j];                 // mix1_w[h][0][j]
    w1b[j] = mix1_w[h * 32 + 16 + j];            // mix1_w[h][1][j]
    b1v[j] = mix1_b[h * 16 + j];
    w2v[j] = mix2_w[h * 16 + j];                 // mix2_w[h][j][0]
  }
  const float b2 = mix2_b[h];

  // Q fragment (A-layout, K padded 16->32: elements 8..15 = 0)
  half16 aQ = {};
  {
    const float* qrow = Q + (size_t)(b * SEQ + r0 + lm) * EMB + h * 16 + g * 8;
#pragma unroll
    for (int e = 0; e < 8; ++e) aQ[e] = (_Float16)qrow[e];
  }

  float8 Oacc = {};
  float mrun[8], lrun[8];
#pragma unroll
  for (int i = 0; i < 8; ++i) { mrun[i] = -1e30f; lrun[i] = 0.f; }

  // cost rows for this lane: base + i*SEQ (compile-time immediate offsets)
  const float* crow = cost + (size_t)b * SEQ * SEQ + (size_t)(r0 + g * 8) * SEQ + lm;
  const _Float16* vrow = Vt + ((size_t)((b * 16 + h) * 16 + lm)) * SEQ;
  const _Float16* krow = Kh + ((size_t)(b * SEQ + lm)) * EMB + h * 16;

  for (int c0 = 0; c0 < SEQ; c0 += 16) {
    // cost tile (independent of the WMMA -> issue first, 8 clause-able loads)
    float cv[8];
#pragma unroll
    for (int i = 0; i < 8; ++i) cv[i] = crow[i * SEQ + c0];

    // K^T fragment (B-layout 32x16; rows 16..31 are the K-dim zero pad)
    half16 bK = {};
    {
      const half8 kz = {};
      const half8* kr = (const half8*)(krow + (size_t)c0 * EMB + g * 8);
      bK = SHUF16(kr[0], kz);
    }
    if (g == 0) {
      if (c0 + 16 < SEQ) {
        __builtin_prefetch(krow + (size_t)(c0 + 16) * EMB, 0, 3);
        __builtin_prefetch(vrow + c0 + 16, 0, 3);
      }
    }
    float8 zero = {};
    float8 s = __builtin_amdgcn_wmma_f32_16x16x32_f16(false, aQ, false, bK,
                                                      (short)0, zero, false, false);

    // mixer MLP: relu(dot*w1a + cost*w1b + b1) . w2 + b2   (per element)
#pragma unroll
    for (int i = 0; i < 8; ++i) {
      const float x = s[i] * 0.25f;              // 1/sqrt(D), D=16
      const float y = cv[i];
      float a2 = b2;
#pragma unroll
      for (int j = 0; j < 16; ++j) {
        float hh = fmaf(x, w1a[j], fmaf(y, w1b[j], b1v[j]));
        a2 = fmaf(fmaxf(hh, 0.f), w2v[j], a2);
      }
      s[i] = a2;
    }

    // online softmax: each row lives across the 16 lanes of a half-wave group
#pragma unroll
    for (int i = 0; i < 8; ++i) {
      float v = s[i];
#pragma unroll
      for (int off = 1; off < 16; off <<= 1)
        v = fmaxf(v, __shfl_xor(v, off, 32));
      const float mnew  = fmaxf(mrun[i], v);
      const float alpha = __expf(mrun[i] - mnew);
      const float p     = __expf(s[i] - mnew);
      float ps = p;
#pragma unroll
      for (int off = 1; off < 16; off <<= 1)
        ps += __shfl_xor(ps, off, 32);
      lrun[i] = lrun[i] * alpha + ps;
      mrun[i] = mnew;
      Oacc[i] *= alpha;
      pl[i + g * 8][lm] = (_Float16)p;           // stage P for A-layout reload
    }
    __syncthreads();
    half16 aP = {};                              // A-layout, K pad -> elems 8..15 = 0
#pragma unroll
    for (int e = 0; e < 8; ++e) aP[e] = pl[lm][g * 8 + e];
    __syncthreads();                             // WAR: next tile rewrites pl

    // V fragment (B-layout; half g: K = g*8 + 0..7, elems 8..15 zero pad); contiguous in Vt
    half16 bV = {};
    {
      const half8 vz = {};
      const half8* vp = (const half8*)(vrow + c0 + g * 8);
      bV = SHUF16(vp[0], vz);
    }
    Oacc = __builtin_amdgcn_wmma_f32_16x16x32_f16(false, aP, false, bV,
                                                  (short)0, Oacc, false, false);
  }

#pragma unroll
  for (int i = 0; i < 8; ++i) {
    const int row = r0 + i + g * 8;
    Out[(size_t)(b * SEQ + row) * EMB + h * 16 + lm] = Oacc[i] / lrun[i];
  }
}

// ---------------------------------------------------------------------------
// InstanceNorm1d over the sequence dim (rows), per (batch, channel), E=256.
// grid (E/32, B), block (32 channels, 8 row-threads).
// ---------------------------------------------------------------------------
__global__ void __launch_bounds__(256)
inorm_kernel(const float* __restrict__ X, const float* __restrict__ gam,
             const float* __restrict__ bet, float* __restrict__ Y, int Nrows)
{
  const int e = blockIdx.x * 32 + threadIdx.x;
  const int b = blockIdx.y;
  const float* x = X + (size_t)b * Nrows * EMB + e;
  float*       y = Y + (size_t)b * Nrows * EMB + e;

  float s = 0.f, s2 = 0.f;
  for (int n = threadIdx.y; n < Nrows; n += 8) {
    const float v = x[(size_t)n * EMB];
    s += v; s2 += v * v;
  }
  __shared__ float sh[2][8][32];
  __shared__ float mu_s[32], rs_s[32];
  sh[0][threadIdx.y][threadIdx.x] = s;
  sh[1][threadIdx.y][threadIdx.x] = s2;
  __syncthreads();
  if (threadIdx.y == 0) {
    float a = 0.f, c = 0.f;
#pragma unroll
    for (int j = 0; j < 8; ++j) { a += sh[0][j][threadIdx.x]; c += sh[1][j][threadIdx.x]; }
    const float mu  = a / (float)Nrows;
    const float var = c / (float)Nrows - mu * mu;
    mu_s[threadIdx.x] = mu;
    rs_s[threadIdx.x] = rsqrtf(var + 1e-5f);
  }
  __syncthreads();
  const float mu = mu_s[threadIdx.x], rs = rs_s[threadIdx.x];
  const float gg = gam[e], bb = bet[e];
  for (int n = threadIdx.y; n < Nrows; n += 8) {
    const float v = x[(size_t)n * EMB];
    y[(size_t)n * EMB] = (v - mu) * rs * gg + bb;
  }
}

// ---------------------------------------------------------------------------
// Host-side orchestration
// ---------------------------------------------------------------------------
static inline void launch_gemm(int mode, const _Float16* A, const _Float16* Bt,
                               const float* bias, const float* Res, float* C,
                               int M, int N, int K, hipStream_t s)
{
  const int tiles = (M / 16) * (N / 16);
  dim3 grid((tiles + 7) / 8), block(256);
  switch (mode) {
    case 0: gemm16_wmma<0><<<grid, block, 0, s>>>(A, Bt, bias, Res, C, M, N, K); break;
    case 1: gemm16_wmma<1><<<grid, block, 0, s>>>(A, Bt, bias, Res, C, M, N, K); break;
    case 2: gemm16_wmma<2><<<grid, block, 0, s>>>(A, Bt, bias, Res, C, M, N, K); break;
    default: gemm16_wmma<3><<<grid, block, 0, s>>>(A, Bt, bias, Res, C, M, N, K); break;
  }
}

static inline dim3 pgrid(int n) { return dim3((n + 255) / 256); }

// Param order (setup_inputs insertion order inside each block dict):
// 0:Wq 1:Wk 2:Wv 3:mix1_w 4:mix1_b 5:mix2_w 6:mix2_b 7:Wo 8:bo
// 9:n1_g 10:n1_b 11:W1 12:b1 13:W2 14:b2 15:n2_g 16:n2_b
static void run_block(const float* xq, const float* xkv, const float* costR,
                      const float* const* P, float* out, float* ws, hipStream_t stream)
{
  const int B = 2, Nq = SEQ, Nk = SEQ;
  const size_t T = (size_t)B * SEQ * EMB;   // 262144 floats = 1 MB
  float* Qb  = ws;                          // f32 Q (read by attention)
  float* Vb  = ws + T;                      // f32 V (pre-transpose)
  float* AO  = ws + 2 * T;                  // attention output (concat heads)
  float* Y1  = ws + 3 * T;                  // pre-norm1
  float* O1  = ws + 4 * T;                  // out1
  float* Hid = ws + 5 * T;                  // FFN hidden (2*T floats)
  float* Y2  = ws + 7 * T;                  // pre-norm2
  _Float16* hbase = (_Float16*)(ws + 8 * T);
  _Float16* Ah  = hbase;                    // packed activations (max 1024x512)
  _Float16* WqT = Ah  + 524288;
  _Float16* WkT = WqT + 65536;
  _Float16* WvT = WkT + 65536;
  _Float16* WoT = WvT + 65536;
  _Float16* W1T = WoT + 65536;
  _Float16* W2T = W1T + 131072;
  _Float16* Khp = W2T + 131072;             // f16 K, (B,C,256) layout
  _Float16* Vtp = Khp + 262144;             // f16 V, (B,H,16,C) layout
  const int Mq = B * Nq;

  // ---- weight pre-pack (f16, transposed) ----
  pack_bt_f16<<<pgrid(256 * 256), 256, 0, stream>>>(P[0],  WqT, 256, 256);
  pack_bt_f16<<<pgrid(256 * 256), 256, 0, stream>>>(P[1],  WkT, 256, 256);
  pack_bt_f16<<<pgrid(256 * 256), 256, 0, stream>>>(P[2],  WvT, 256, 256);
  pack_bt_f16<<<pgrid(256 * 256), 256, 0, stream>>>(P[7],  WoT, 256, 256);
  pack_bt_f16<<<pgrid(256 * 512), 256, 0, stream>>>(P[11], W1T, 256, 512);
  pack_bt_f16<<<pgrid(512 * 256), 256, 0, stream>>>(P[13], W2T, 512, 256);

  // ---- QKV projections ----
  pack_f16<<<pgrid(T), 256, 0, stream>>>(xq, Ah, (int)T);
  launch_gemm(0, Ah, WqT, nullptr, nullptr, Qb, Mq, 256, 256, stream);
  pack_f16<<<pgrid(T), 256, 0, stream>>>(xkv, Ah, (int)T);
  launch_gemm(3, Ah, WkT, nullptr, nullptr, (float*)Khp, Mq, 256, 256, stream); // f16 out
  launch_gemm(0, Ah, WvT, nullptr, nullptr, Vb, Mq, 256, 256, stream);
  pack_vt_f16<<<pgrid(2 * Nk * 256), 256, 0, stream>>>(Vb, Vtp, Nk);

  // ---- fused mixed-score flash attention ----
  dim3 agrid(B * 16 * (Nq / 16)), ablock(32);
  attn_mixed_flash<<<agrid, ablock, 0, stream>>>(
      Qb, Khp, Vtp, costR, P[3], P[4], P[5], P[6], AO);

  // ---- output projection + residual + norm1 ----
  pack_f16<<<pgrid(T), 256, 0, stream>>>(AO, Ah, (int)T);
  launch_gemm(2, Ah, WoT, P[8], xq, Y1, Mq, 256, 256, stream);
  dim3 ngrid(EMB / 32, B), nblock(32, 8);
  inorm_kernel<<<ngrid, nblock, 0, stream>>>(Y1, P[9], P[10], O1, Nq);

  // ---- FFN + residual + norm2 ----
  pack_f16<<<pgrid(T), 256, 0, stream>>>(O1, Ah, (int)T);
  launch_gemm(1, Ah, W1T, P[12], nullptr, Hid, Mq, 512, 256, stream);
  pack_f16<<<pgrid(2 * T), 256, 0, stream>>>(Hid, Ah, (int)(2 * T));
  launch_gemm(2, Ah, W2T, P[14], O1, Y2, Mq, 256, 512, stream);
  inorm_kernel<<<ngrid, nblock, 0, stream>>>(Y2, P[15], P[16], out, Nq);
}

extern "C" void kernel_launch(void* const* d_in, const int* in_sizes, int n_in,
                              void* d_out, int out_size, void* d_ws, size_t ws_size,
                              hipStream_t stream)
{
  (void)in_sizes; (void)n_in; (void)out_size; (void)ws_size;
  const float* row_emb = (const float*)d_in[0];
  const float* col_emb = (const float*)d_in[1];
  const float* cost    = (const float*)d_in[2];
  const float* Prow[17];
  const float* Pcol[17];
  for (int i = 0; i < 17; ++i) {
    Prow[i] = (const float*)d_in[3 + i];
    Pcol[i] = (const float*)d_in[20 + i];
  }
  float* ws  = (float*)d_ws;
  float* out = (float*)d_out;

  // cost^T staged once (after the per-block scratch region)
  const size_t T = (size_t)2 * SEQ * EMB;
  float* costT = ws + 8 * T + 786432 / 1;   // halves region = 1572864 h = 786432 f
  // (ws + 8T .. +786432 floats) is the f16 packing area; costT goes after it.
  costT = ws + 8 * T + 786432;

  transpose512<<<dim3(SEQ / 32, SEQ / 32, 2), 256, 0, stream>>>(cost, costT);

  // row block: queries = row_emb, keys/values = col_emb, cost row-major
  run_block(row_emb, col_emb, cost, Prow, out, ws, stream);
  // col block: queries = col_emb, keys/values = row_emb, cost transposed
  run_block(col_emb, row_emb, costT, Pcol, out + T, ws, stream);
}
